// StreamingAttention_82197084111458
// MI455X (gfx1250) — compile-verified, matrix operand order checked by reading, not run on hardware
//
#include <hip/hip_runtime.h>
#include <hip/hip_bf16.h>
#include <math.h>

#define D_MODEL 1024
#define N_HEADS 16
#define HEAD_DIM 64
#define BATCH 4
#define SEQ 2048

typedef __attribute__((ext_vector_type(16))) __bf16 v16bf;
typedef __attribute__((ext_vector_type(8)))  float  v8f;

// Load a 16x32 bf16 WMMA fragment (A-layout, or B-layout when memory is
// [N][K] with K contiguous). ISA 7.12.2: element e -> K = (e/8)*16 +
// (lane/16)*8 + (e%8). Elements 0..7 and 8..15 are each 8 contiguous bf16
// (16 bytes) => two B128 loads per fragment.
static __device__ __forceinline__ v16bf load_frag16(const __bf16* base,
                                                    int row_stride, int row, int k0) {
    const int lane = threadIdx.x & 31;
    const int hk = (lane >> 4) * 8;
    const __bf16* p = base + (size_t)row * row_stride + k0 + hk;
    union { v16bf v; uint4 q[2]; } f;
    f.q[0] = *(const uint4*)(p);
    f.q[1] = *(const uint4*)(p + 16);
    return f.v;
}

static __device__ __forceinline__ v8f wmma_bf16(v16bf a, v16bf b, v8f c) {
    return __builtin_amdgcn_wmma_f32_16x16x32_bf16(false, a, false, b,
                                                   (short)0, c, false, false);
}

// vectorized fp32 -> bf16 LDS staging (float4 in, 8-byte store out)
static __device__ __forceinline__ void stage_bf16(const float* __restrict__ src,
                                                  __bf16* dst, int total, int tid, int nthreads) {
    for (int i = tid * 4; i < total; i += nthreads * 4) {
        float4 f = *(const float4*)(src + i);
        union { __bf16 b[4]; unsigned u[2]; } w;
        w.b[0] = (__bf16)f.x; w.b[1] = (__bf16)f.y;
        w.b[2] = (__bf16)f.z; w.b[3] = (__bf16)f.w;
        *(uint2*)(dst + i) = make_uint2(w.u[0], w.u[1]);
    }
}

// ISA 10.2: flat/generic address in the LDS aperture maps to LDS byte
// address by truncation to 32 bits.
static __device__ __forceinline__ unsigned lds_off(const void* p) {
    return (unsigned)(uintptr_t)p;
}

// Async 16-byte copy global -> LDS (GVS mode), tracked by ASYNCcnt.
static __device__ __forceinline__ void async_copy_b128(unsigned lds_addr,
                                                       unsigned goff,
                                                       unsigned long long gbase) {
    asm volatile("global_load_async_to_lds_b128 %0, %1, %2"
                 :: "v"(lds_addr), "v"(goff), "s"(gbase) : "memory");
}

// =====================================================================
// Kernel 0: one-shot weight convert + transpose to bf16.
//   WqkvT: [z][h][n(64)][k(1024)]   WoT: [n(1024)][k(1024)]
// =====================================================================
#define QKV_W_TOTAL (3 * N_HEADS * D_MODEL * HEAD_DIM)
#define WO_TOTAL    (D_MODEL * D_MODEL)

__global__ void __launch_bounds__(256)
convert_weights_kernel(const float* __restrict__ Wq, const float* __restrict__ Wk,
                       const float* __restrict__ Wv, const float* __restrict__ Wo,
                       __bf16* __restrict__ WqkvT, __bf16* __restrict__ WoT) {
    size_t i = (size_t)blockIdx.x * blockDim.x + threadIdx.x;
    if (i < QKV_W_TOTAL) {
        int z  = (int)(i / (N_HEADS * D_MODEL * HEAD_DIM));
        int r  = (int)(i % (N_HEADS * D_MODEL * HEAD_DIM));
        int h  = r / (D_MODEL * HEAD_DIM);
        int r2 = r % (D_MODEL * HEAD_DIM);
        int k  = r2 / HEAD_DIM;
        int n  = r2 % HEAD_DIM;
        const float* W = (z == 0 ? Wq : z == 1 ? Wk : Wv);
        WqkvT[(((size_t)z * N_HEADS + h) * HEAD_DIM + n) * D_MODEL + k] =
            (__bf16)W[(size_t)h * D_MODEL * HEAD_DIM + r2];
    } else if (i < (size_t)QKV_W_TOTAL + WO_TOTAL) {
        int j = (int)(i - QKV_W_TOTAL);
        int k = j / D_MODEL;
        int n = j % D_MODEL;
        WoT[(size_t)n * D_MODEL + k] = (__bf16)Wo[j];
    }
}

// =====================================================================
// Kernel 1: QKV projection + bias + RoPE.
//   grid = (B*L/16, H, 3[q,k,v]) , block = 128 (4 waves)
//   Q,K row-major bf16 [(b*H+h)*L + l][64]; V stored TRANSPOSED bf16
//   [(b*H+h)*64 + e][L] so PV B-fragments load contiguously.
// =====================================================================
__global__ void __launch_bounds__(128)
qkv_rope_kernel(const float* __restrict__ x, const __bf16* __restrict__ WqkvT,
                const float* __restrict__ bq, const float* __restrict__ bk,
                const float* __restrict__ bv,
                __bf16* __restrict__ Qo, __bf16* __restrict__ Ko,
                __bf16* __restrict__ Vt) {
    __shared__ alignas(16) __bf16 a_lds[16][D_MODEL];   // 32 KB row strip of x
    __shared__ alignas(16) float  out_lds[16][HEAD_DIM];

    const int tile = blockIdx.x;
    const int h    = blockIdx.y;
    const int z    = blockIdx.z;           // 0=q 1=k 2=v
    const int wid  = threadIdx.x >> 5;
    const int lane = threadIdx.x & 31;

    const float* xb = x + (size_t)tile * 16 * D_MODEL;
    stage_bf16(xb, &a_lds[0][0], 16 * D_MODEL, threadIdx.x, 128);
    __syncthreads();

    const __bf16* Wt   = WqkvT + ((size_t)z * N_HEADS + h) * HEAD_DIM * D_MODEL;
    const float*  bias = (z == 0 ? bq : z == 1 ? bk : bv) + (size_t)h * HEAD_DIM;

    v8f acc = {0.f, 0.f, 0.f, 0.f, 0.f, 0.f, 0.f, 0.f};
    const int ncol = wid * 16 + (lane & 15);
#pragma unroll 4
    for (int kk = 0; kk < D_MODEL; kk += 32) {
        v16bf a = load_frag16(&a_lds[0][0], D_MODEL, lane & 15, kk);
        v16bf b = load_frag16(Wt, D_MODEL, ncol, kk);     // [n][k] bf16
        acc = wmma_bf16(a, b, acc);
    }
    const int hi = lane >> 4;
#pragma unroll
    for (int j = 0; j < 8; ++j)
        out_lds[j + 8 * hi][ncol] = acc[j];
    __syncthreads();

    // epilogue: bias + RoPE (pairs e, e^32) + store bf16
    for (int i = threadIdx.x; i < 16 * HEAD_DIM; i += 128) {
        int r = i >> 6, e = i & 63;
        int grow = tile * 16 + r;
        int l = grow & (SEQ - 1);
        int bix = grow >> 11;
        float val = out_lds[r][e] + bias[e];
        if (z < 2) {
            int eh = e & 31;
            float inv = __powf(10000.f, -(float)eh * (1.f / 32.f));
            float sn, cs;
            __sincosf((float)l * inv, &sn, &cs);
            float other = out_lds[r][e ^ 32] + bias[e ^ 32];
            val = (e < 32) ? (val * cs - other * sn) : (other * sn + val * cs);
        }
        __bf16 bf = (__bf16)val;
        size_t bh = ((size_t)bix * N_HEADS + h);
        if (z == 0)      Qo[(bh * SEQ + l) * HEAD_DIM + e] = bf;
        else if (z == 1) Ko[(bh * SEQ + l) * HEAD_DIM + e] = bf;
        else             Vt[(bh * HEAD_DIM + e) * SEQ + l] = bf;
    }
}

// =====================================================================
// Kernel 2: flash attention (non-causal, full L).
//   grid = (L/64, H, B), block = 128 (4 waves); wave w owns q-tile
//   blockIdx.x*4+w. K/V chunks (32 keys) are async-staged into
//   double-buffered LDS shared by all 4 waves (4x traffic reduction),
//   pipelined with s_wait_asynccnt.
// =====================================================================
__global__ void __launch_bounds__(128)
flash_attn_kernel(const __bf16* __restrict__ Qb, const __bf16* __restrict__ Kb,
                  const __bf16* __restrict__ Vt, float* __restrict__ ctx) {
    __shared__ alignas(16) __bf16 k_lds[2][32][HEAD_DIM];  // 2 x 4 KB [key][hd]
    __shared__ alignas(16) __bf16 v_lds[2][HEAD_DIM][32];  // 2 x 4 KB [col][key]
    __shared__ alignas(16) __bf16 p_lds[4][16][32];        // per-wave P tile

    const int tid  = threadIdx.x;
    const int wid  = tid >> 5;
    const int lane = tid & 31;
    const int hi   = lane >> 4;
    const int ln16 = lane & 15;
    const int qtile = blockIdx.x * 4 + wid;
    const size_t bh = (size_t)blockIdx.z * N_HEADS + blockIdx.y;

    const __bf16* Qp = Qb + (bh * SEQ + (size_t)qtile * 16) * HEAD_DIM;
    const unsigned long long Kbase = (unsigned long long)(Kb + bh * SEQ * HEAD_DIM);
    const unsigned long long Vbase = (unsigned long long)(Vt + bh * (size_t)HEAD_DIM * SEQ);

    v16bf aq0 = load_frag16(Qp, HEAD_DIM, ln16, 0);
    v16bf aq1 = load_frag16(Qp, HEAD_DIM, ln16, 32);

    float m[8], lsum[8];
    v8f o[4];
#pragma unroll
    for (int j = 0; j < 8; ++j) { m[j] = -1e30f; lsum[j] = 0.f; }
#pragma unroll
    for (int t = 0; t < 4; ++t)
        o[t] = (v8f){0.f, 0.f, 0.f, 0.f, 0.f, 0.f, 0.f, 0.f};

    // issue async staging of one 32-key chunk (4 b128 per thread:
    // 2 for K [32x64 bf16 = 4KB contiguous], 2 for V^T [64 rows x 64B]).
    auto issue_chunk = [&](int kc, int bb) {
        unsigned kl = lds_off(&k_lds[bb][0][0]);
        unsigned vl = lds_off(&v_lds[bb][0][0]);
#pragma unroll
        for (int i = 0; i < 2; ++i) {
            int idx = tid + i * 128;                       // 0..255
            async_copy_b128(kl + idx * 16, (unsigned)(kc * 128 + idx * 16), Kbase);
        }
#pragma unroll
        for (int i = 0; i < 2; ++i) {
            int idx = tid + i * 128;                       // 0..255
            int col = idx >> 2, j = idx & 3;               // 4 x 16B per 64B row
            async_copy_b128(vl + idx * 16,
                            (unsigned)(col * (SEQ * 2) + kc * 2 + j * 16), Vbase);
        }
    };

    const int NCH = SEQ / 32;
    issue_chunk(0, 0);
    for (int ic = 0; ic < NCH; ++ic) {
        const int bb = ic & 1;
        if (ic + 1 < NCH) {
            issue_chunk((ic + 1) * 32, bb ^ 1);
            asm volatile("s_wait_asynccnt 4" ::: "memory");  // chunk ic landed
        } else {
            asm volatile("s_wait_asynccnt 0" ::: "memory");
        }
        __syncthreads();            // all waves' shares of buffer bb visible

        v8f s0 = {0.f,0.f,0.f,0.f,0.f,0.f,0.f,0.f};
        v8f s1 = s0;
        const __bf16* kp = &k_lds[bb][0][0];
        {
            v16bf b0a = load_frag16(kp, HEAD_DIM, ln16, 0);
            v16bf b0b = load_frag16(kp, HEAD_DIM, ln16, 32);
            s0 = wmma_bf16(aq0, b0a, s0);
            s0 = wmma_bf16(aq1, b0b, s0);
            v16bf b1a = load_frag16(kp, HEAD_DIM, 16 + ln16, 0);
            v16bf b1b = load_frag16(kp, HEAD_DIM, 16 + ln16, 32);
            s1 = wmma_bf16(aq0, b1a, s1);
            s1 = wmma_bf16(aq1, b1b, s1);
        }
        // online softmax; C-fragment row j lives on a 16-lane group
#pragma unroll
        for (int j = 0; j < 8; ++j) {
            float v0 = s0[j] * 0.125f;           // 1/sqrt(64)
            float v1 = s1[j] * 0.125f;
            float rm = fmaxf(v0, v1);
#pragma unroll
            for (int d = 1; d < 16; d <<= 1) rm = fmaxf(rm, __shfl_xor(rm, d, 16));
            float mn = fmaxf(m[j], rm);
            float alpha = __expf(m[j] - mn);
            float p0 = __expf(v0 - mn);
            float p1 = __expf(v1 - mn);
            float rs = p0 + p1;
#pragma unroll
            for (int d = 1; d < 16; d <<= 1) rs += __shfl_xor(rs, d, 16);
            lsum[j] = lsum[j] * alpha + rs;
            m[j] = mn;
#pragma unroll
            for (int t = 0; t < 4; ++t) o[t][j] *= alpha;
            int row = j + 8 * hi;
            p_lds[wid][row][ln16]      = (__bf16)p0;
            p_lds[wid][row][16 + ln16] = (__bf16)p1;
        }
        // p_lds is wave-private; per-wave DS ops are in-order -> fence only
        asm volatile("" ::: "memory");
        v16bf pa = load_frag16(&p_lds[0][0][0] + (size_t)wid * 16 * 32, 32, ln16, 0);
        const __bf16* vp = &v_lds[bb][0][0];
#pragma unroll
        for (int t = 0; t < 4; ++t) {
            v16bf bv = load_frag16(vp, 32, t * 16 + ln16, 0);
            o[t] = wmma_bf16(pa, bv, o[t]);
        }
        __syncthreads();            // everyone done with buffer bb
    }

    // normalize and scatter to merged ctx (B, L, H*64) fp32
#pragma unroll
    for (int t = 0; t < 4; ++t) {
#pragma unroll
        for (int j = 0; j < 8; ++j) {
            int row = qtile * 16 + j + 8 * hi;
            int col = blockIdx.y * HEAD_DIM + t * 16 + ln16;
            ctx[((size_t)blockIdx.z * SEQ + row) * D_MODEL + col] = o[t][j] / lsum[j];
        }
    }
}

// =====================================================================
// Kernel 3: output projection  out = ctx @ Wo + bo
//   grid = (B*L/16, 8), block = 256 (8 waves, 128 output cols per block)
// =====================================================================
__global__ void __launch_bounds__(256)
out_proj_kernel(const float* __restrict__ ctx, const __bf16* __restrict__ WoT,
                const float* __restrict__ bo, float* __restrict__ out) {
    __shared__ alignas(16) __bf16 a_lds[16][D_MODEL];   // 32 KB

    const int tile = blockIdx.x;
    const int wid  = threadIdx.x >> 5;
    const int lane = threadIdx.x & 31;

    stage_bf16(ctx + (size_t)tile * 16 * D_MODEL, &a_lds[0][0],
               16 * D_MODEL, threadIdx.x, 256);
    __syncthreads();

    const int ncol = (blockIdx.y * 8 + wid) * 16 + (lane & 15);
    v8f acc = {0.f, 0.f, 0.f, 0.f, 0.f, 0.f, 0.f, 0.f};
#pragma unroll 4
    for (int kk = 0; kk < D_MODEL; kk += 32) {
        v16bf a = load_frag16(&a_lds[0][0], D_MODEL, lane & 15, kk);
        v16bf b = load_frag16(WoT, D_MODEL, ncol, kk);    // [n][k] bf16
        acc = wmma_bf16(a, b, acc);
    }
    const int hi = lane >> 4;
    const float bias = bo[ncol];
#pragma unroll
    for (int j = 0; j < 8; ++j) {
        int row = tile * 16 + j + 8 * hi;
        out[(size_t)row * D_MODEL + ncol] = acc[j] + bias;
    }
}

// =====================================================================
extern "C" void kernel_launch(void* const* d_in, const int* in_sizes, int n_in,
                              void* d_out, int out_size, void* d_ws, size_t ws_size,
                              hipStream_t stream) {
    const float* x  = (const float*)d_in[0];
    const float* Wq = (const float*)d_in[1];
    const float* bq = (const float*)d_in[2];
    const float* Wk = (const float*)d_in[3];
    const float* bk = (const float*)d_in[4];
    const float* Wv = (const float*)d_in[5];
    const float* bv = (const float*)d_in[6];
    const float* Wo = (const float*)d_in[7];
    const float* bo = (const float*)d_in[8];
    float* out = (float*)d_out;

    // workspace: Q | K | V^T (bf16) | ctx (fp32) | WqkvT | WoT   ~ 92 MB
    const size_t nQKV = (size_t)BATCH * N_HEADS * SEQ * HEAD_DIM;   // 8.39M elems
    __bf16* Qw = (__bf16*)d_ws;
    __bf16* Kw = Qw + nQKV;
    __bf16* Vw = Kw + nQKV;
    float* ctx = (float*)(Vw + nQKV);
    __bf16* WqkvT = (__bf16*)(ctx + (size_t)BATCH * SEQ * D_MODEL);
    __bf16* WoT   = WqkvT + (size_t)QKV_W_TOTAL;

    const int wtot = QKV_W_TOTAL + WO_TOTAL;
    convert_weights_kernel<<<(wtot + 255) / 256, 256, 0, stream>>>(Wq, Wk, Wv, Wo,
                                                                  WqkvT, WoT);

    dim3 g1(BATCH * SEQ / 16, N_HEADS, 3);
    qkv_rope_kernel<<<g1, 128, 0, stream>>>(x, WqkvT, bq, bk, bv, Qw, Kw, Vw);

    dim3 g2(SEQ / 64, N_HEADS, BATCH);
    flash_attn_kernel<<<g2, 128, 0, stream>>>(Qw, Kw, Vw, ctx);

    dim3 g3(BATCH * SEQ / 16, 8);
    out_proj_kernel<<<g3, 256, 0, stream>>>(ctx, WoT, bo, out);
}
